// DeformableTransformer_48996986912889
// MI455X (gfx1250) — matrix operand order repaired and checked by reading, not compiled, not run on hardware
//
#include <hip/hip_runtime.h>

// ---------------- static problem config (matches reference) ----------------
#define N_B   8
#define LQ    300
#define DM    256
#define NH    8
#define HD    32
#define L_LV  4
#define P_PT  4
#define FFH   1024
#define LYR   6
#define S_IN  21760
#define M_Q   (N_B * LQ)     // 2400
#define M_V   (N_B * S_IN)   // 174080

typedef __attribute__((ext_vector_type(16))) _Float16 v16h;
typedef __attribute__((ext_vector_type(8)))  _Float16 v8h;
typedef __attribute__((ext_vector_type(4)))  _Float16 v4h;
typedef __attribute__((ext_vector_type(8)))  float    v8f;
typedef __attribute__((ext_vector_type(4)))  float    v4f;

// ---------------- WMMA GEMM:  C[M,Nout] = A[M,K] * W[Nout,K]^T + bias -------
// A, W f32 row-major (K contiguous), converted to packed f16 at load time.
// Tile: 128(M) x 64(N) x 64(K), double-buffered LDS, 1 barrier / 8 WMMAs.
// 256 threads = 8 waves; each wave owns a 32x32 sub-tile = 2x2 WMMA, 2 K-substeps.
// Row indices are CLAMPED (not predicated): an A row >= M only feeds output
// rows >= M, which are never stored, so unconditional loads are safe.
#define BM 128
#define BN 64
#define BK 64
#define LDSK 72   // padded row stride in halves (144B: 8B aligned, bank-spread)

__global__ __launch_bounds__(256)
void gemm_wmma_kernel(const float* __restrict__ A, const float* __restrict__ W,
                      const float* __restrict__ bias, float* __restrict__ C,
                      int M, int Nout, int K, int ldc, int relu, int half_out)
{
    __shared__ _Float16 As[2][BM * LDSK];
    __shared__ _Float16 Bs[2][BN * LDSK];

    const int tid  = threadIdx.x;
    const int lane = tid & 31;
    const int wave = tid >> 5;
    const int m0 = blockIdx.y * BM;
    const int n0 = blockIdx.x * BN;
    const int wm = (wave & 3) * 32;   // wave's M offset within tile
    const int wn = (wave >> 2) * 32;  // wave's N offset within tile

    // staging pattern: thread covers (row = it*16 + tid/16, col = (tid%16)*4)
    const int lr = tid >> 4;
    const int lc = (tid & 15) * 4;

    const float* aptr[8];
    #pragma unroll
    for (int it = 0; it < 8; ++it) {
        int gm = m0 + it * 16 + lr;
        if (gm > M - 1) gm = M - 1;              // clamp, never predicate
        aptr[it] = A + (size_t)gm * K + lc;
    }
    const float* wptr[4];
    #pragma unroll
    for (int it = 0; it < 4; ++it) {
        int gn = n0 + it * 16 + lr;
        if (gn > Nout - 1) gn = Nout - 1;        // clamp
        wptr[it] = W + (size_t)gn * K + lc;
    }

    // staged data kept as packed f16 (2 VGPRs/chunk) to minimize live pressure
    v4h ra[8], rw[4];
    v8f acc[2][2] = {};

    auto loadregs = [&]() {
        #pragma unroll
        for (int it = 0; it < 8; ++it) {
            v4f v = *(const v4f*)aptr[it];
            aptr[it] += BK;                      // in-place bump, no temp quads
            v4h h;
            h[0] = (_Float16)v[0]; h[1] = (_Float16)v[1];
            h[2] = (_Float16)v[2]; h[3] = (_Float16)v[3];
            ra[it] = h;
        }
        #pragma unroll
        for (int it = 0; it < 4; ++it) {
            v4f v = *(const v4f*)wptr[it];
            wptr[it] += BK;
            v4h h;
            h[0] = (_Float16)v[0]; h[1] = (_Float16)v[1];
            h[2] = (_Float16)v[2]; h[3] = (_Float16)v[3];
            rw[it] = h;
        }
    };
    auto storeregs = [&](int buf) {
        #pragma unroll
        for (int it = 0; it < 8; ++it)
            *(v4h*)&As[buf][(it * 16 + lr) * LDSK + lc] = ra[it];   // ds_store_b64
        #pragma unroll
        for (int it = 0; it < 4; ++it)
            *(v4h*)&Bs[buf][(it * 16 + lr) * LDSK + lc] = rw[it];
    };
    // fragment layouts per ISA 7.12.2:
    // A (16x32, f16): lane L -> row L%16; halves 0..7 = K (L<16?0:8)+i,
    //                 halves 8..15 = K (L<16?16:24)+i
    // B (32x16):      lane L -> col L%16; same K mapping as A
    auto compute = [&](int buf) {
        #pragma unroll
        for (int ks = 0; ks < 2; ++ks) {
            v16h af[2], bf[2];
            const int arow = wm + (lane & 15);
            const int ab0  = ((lane < 16) ? 0 : 8) + ks * 32;
            #pragma unroll
            for (int i = 0; i < 2; ++i) {
                const _Float16* p = &As[buf][(arow + i * 16) * LDSK];
                v8h lo = *(const v8h*)(p + ab0);
                v8h hi = *(const v8h*)(p + ab0 + 16);
                af[i] = __builtin_shufflevector(lo, hi,
                        0,1,2,3,4,5,6,7,8,9,10,11,12,13,14,15);
            }
            const int bcol = wn + (lane & 15);
            const int bk0  = ((lane < 16) ? 0 : 8) + ks * 32;
            #pragma unroll
            for (int j = 0; j < 2; ++j) {
                const _Float16* p = &Bs[buf][(bcol + j * 16) * LDSK] + bk0;
                v8h lo = *(const v8h*)(p);
                v8h hi = *(const v8h*)(p + 16);
                bf[j] = __builtin_shufflevector(lo, hi,
                        0,1,2,3,4,5,6,7,8,9,10,11,12,13,14,15);
            }
            #pragma unroll
            for (int i = 0; i < 2; ++i)
                #pragma unroll
                for (int j = 0; j < 2; ++j)
                    acc[i][j] = __builtin_amdgcn_wmma_f32_16x16x32_f16(
                        false, af[i], false, bf[j], (short)0, acc[i][j],
                        false, false);
        }
    };

    // software pipeline: global loads for tile k+1 issue before WMMAs of tile k
    loadregs();
    storeregs(0);
    __syncthreads();
    int buf = 0;
    for (int k0 = BK; k0 < K; k0 += BK) {
        loadregs();            // global_load_b128 x12 + cvt_pk (next tile)
        compute(buf);          // ds_load + 8 x v_wmma (current tile)
        storeregs(buf ^ 1);    // ds_store_b64 x12 (next tile)
        __syncthreads();       // single barrier per K-step
        buf ^= 1;
    }
    compute(buf);

    // epilogue: C/D layout — VGPR v holds M = v (lanes 0-15) / v+8 (lanes 16-31)
    const int mo = (lane < 16) ? 0 : 8;
    #pragma unroll
    for (int i = 0; i < 2; ++i) {
        #pragma unroll
        for (int j = 0; j < 2; ++j) {
            int col = n0 + wn + j * 16 + (lane & 15);
            if (col >= Nout) continue;
            float bv = bias ? bias[col] : 0.f;
            #pragma unroll
            for (int v = 0; v < 8; ++v) {
                int row = m0 + wm + i * 16 + mo + v;
                if (row < M) {
                    float r = acc[i][j][v] + bv;
                    if (relu) r = fmaxf(r, 0.f);
                    if (half_out) ((_Float16*)C)[(size_t)row * ldc + col] = (_Float16)r;
                    else          C[(size_t)row * ldc + col] = r;
                }
            }
        }
    }
}

// ---------------- elementwise add ----------------
__global__ void add_kernel(const float* __restrict__ a, const float* __restrict__ b,
                           float* __restrict__ o, int n)
{
    int i = blockIdx.x * blockDim.x + threadIdx.x;
    if (i < n) o[i] = a[i] + b[i];
}

__global__ void copy_kernel(const float* __restrict__ a, float* __restrict__ o, int n)
{
    int i = blockIdx.x * blockDim.x + threadIdx.x;
    if (i < n) o[i] = a[i];
}

// ---------------- self-attention: softmax(QK^T/sqrt(HD)) V  per (b,h,q) -----
// qkv layout: (N,LQ,768): cols 0..255 q, 256..511 k, 512..767 v
__global__ __launch_bounds__(128)
void self_attn_kernel(const float* __restrict__ qkv, float* __restrict__ out)
{
    const int q = blockIdx.x, h = blockIdx.y, b = blockIdx.z;
    const int t = threadIdx.x;
    __shared__ float sl[LQ];
    __shared__ float red[128];
    __shared__ float qv[HD];
    if (t < HD) qv[t] = qkv[((size_t)(b * LQ + q)) * 768 + h * HD + t];
    __syncthreads();
    const float scale = 0.17677669529663687f;  // 1/sqrt(32)
    for (int k = t; k < LQ; k += 128) {
        const float* kr = qkv + ((size_t)(b * LQ + k)) * 768 + DM + h * HD;
        float d = 0.f;
        #pragma unroll
        for (int i = 0; i < HD; ++i) d += qv[i] * kr[i];
        sl[k] = d * scale;
    }
    __syncthreads();
    float m = -3.0e38f;
    for (int k = t; k < LQ; k += 128) m = fmaxf(m, sl[k]);
    red[t] = m; __syncthreads();
    for (int s = 64; s > 0; s >>= 1) { if (t < s) red[t] = fmaxf(red[t], red[t + s]); __syncthreads(); }
    m = red[0];
    __syncthreads();
    float lsum = 0.f;
    for (int k = t; k < LQ; k += 128) { float e = __expf(sl[k] - m); sl[k] = e; lsum += e; }
    red[t] = lsum; __syncthreads();
    for (int s = 64; s > 0; s >>= 1) { if (t < s) red[t] += red[t + s]; __syncthreads(); }
    float inv = 1.f / red[0];
    if (t < HD) {
        float a = 0.f;
        const float* vb = qkv + (size_t)(b * LQ) * 768 + 2 * DM + h * HD + t;
        for (int k = 0; k < LQ; ++k) a += sl[k] * vb[(size_t)k * 768];
        out[((size_t)(b * LQ + q)) * DM + h * HD + t] = a * inv;
    }
}

// ---------------- residual add + LayerNorm over D=256 ----------------
__global__ __launch_bounds__(256)
void add_ln_kernel(const float* __restrict__ x, const float* __restrict__ dlt,
                   const float* __restrict__ g, const float* __restrict__ bta,
                   float* __restrict__ out)
{
    const int row = blockIdx.x, t = threadIdx.x;
    __shared__ float red[256];
    float v = x[(size_t)row * DM + t] + dlt[(size_t)row * DM + t];
    red[t] = v; __syncthreads();
    for (int s = 128; s > 0; s >>= 1) { if (t < s) red[t] += red[t + s]; __syncthreads(); }
    float mean = red[0] * (1.f / DM);
    __syncthreads();
    float c = v - mean;
    red[t] = c * c; __syncthreads();
    for (int s = 128; s > 0; s >>= 1) { if (t < s) red[t] += red[t + s]; __syncthreads(); }
    float var = red[0] * (1.f / DM);
    out[(size_t)row * DM + t] = c * rsqrtf(var + 1e-5f) * g[t] + bta[t];
}

// ---------------- softmax over L*P=16 per (row, head) on 128-wide rows ------
__global__ __launch_bounds__(128)
void aw_softmax_kernel(float* __restrict__ aw)
{
    const int row = blockIdx.x, t = threadIdx.x;
    __shared__ float s[128];
    float v = aw[(size_t)row * 128 + t];
    s[t] = v; __syncthreads();
    int g0 = t & ~15;
    float m = -3.0e38f;
    #pragma unroll
    for (int i = 0; i < 16; ++i) m = fmaxf(m, s[g0 + i]);
    float e = __expf(v - m);
    __syncthreads(); s[t] = e; __syncthreads();
    float sum = 0.f;
    #pragma unroll
    for (int i = 0; i < 16; ++i) sum += s[g0 + i];
    aw[(size_t)row * 128 + t] = e / sum;
}

// ---------------- sampling locations: ref*valid + off/norm ----------------
// layout per row (n,q): t = h*32 + l*8 + p*2 + c
__global__ __launch_bounds__(256)
void loc_kernel(const float* __restrict__ off, const float* __restrict__ refp,
                const float* __restrict__ svr, float* __restrict__ loc)
{
    const int row = blockIdx.x;
    const int b = row / LQ;
    const int t = threadIdx.x;
    const int c = t & 1;
    const int l = (t >> 3) & 3;
    float nrm = (float)(128 >> l);  // W==H per level: 128,64,32,16
    float ref = refp[(size_t)row * 2 + c] * svr[(b * L_LV + l) * 2 + c];
    loc[(size_t)row * 256 + t] = ref + off[(size_t)row * 256 + t] / nrm;
}

// ---------------- deformable bilinear sampling + weighted sum --------------
// value: (N, S_IN, 256) f16; out: (N*LQ, 256) f32. thread t: h=t/32, d=t%32
__global__ __launch_bounds__(256)
void deform_sample_kernel(const _Float16* __restrict__ value, const float* __restrict__ loc,
                          const float* __restrict__ aw, float* __restrict__ out)
{
    const int row = blockIdx.x;
    const int b = row / LQ;
    const int t = threadIdx.x;
    const int h = t >> 5, d = t & 31;
    __shared__ float sloc[256];
    __shared__ float saw[128];
    sloc[t] = loc[(size_t)row * 256 + t];
    if (t < 128) saw[t] = aw[(size_t)row * 128 + t];
    __syncthreads();
    const int S0[4] = {0, 16384, 20480, 21504};
    float acc = 0.f;
    #pragma unroll
    for (int l = 0; l < 4; ++l) {
        const int hw = 128 >> l;  // H_ == W_
        const int s0 = S0[l];
        #pragma unroll
        for (int p = 0; p < 4; ++p) {
            float x = sloc[h * 32 + l * 8 + p * 2 + 0] * hw - 0.5f;
            float y = sloc[h * 32 + l * 8 + p * 2 + 1] * hw - 0.5f;
            float x0f = floorf(x), y0f = floorf(y);
            float lx = x - x0f, ly = y - y0f;
            int x0 = (int)x0f, y0 = (int)y0f;
            float a = saw[h * 16 + l * 4 + p];
            float tap[4];
            #pragma unroll
            for (int q2 = 0; q2 < 4; ++q2) {
                int yy = y0 + (q2 >> 1), xx = x0 + (q2 & 1);
                float vv = 0.f;
                if (yy >= 0 && yy < hw && xx >= 0 && xx < hw) {
                    size_t idx = ((size_t)b * S_IN + s0 + (size_t)yy * hw + xx) * DM + h * HD + d;
                    vv = (float)value[idx];
                }
                tap[q2] = vv;
            }
            acc += a * ((1.f - ly) * ((1.f - lx) * tap[0] + lx * tap[1]) +
                        ly         * ((1.f - lx) * tap[2] + lx * tap[3]));
        }
    }
    out[(size_t)row * DM + t] = acc;
}

// ---------------- top-30 of 128 weights, gather sampling points ------------
__global__ __launch_bounds__(128)
void topk_kernel(const float* __restrict__ aw, const float* __restrict__ loc,
                 const float* __restrict__ svr, float* __restrict__ out)
{
    const int row = blockIdx.x;
    const int b = row / LQ;
    const int t = threadIdx.x;
    __shared__ float w[128];
    __shared__ float bw[128];
    __shared__ int   bidx[128];
    w[t] = aw[(size_t)row * 128 + t];
    __syncthreads();
    for (int j = 0; j < 30; ++j) {
        bw[t] = w[t]; bidx[t] = t;
        __syncthreads();
        for (int s = 64; s > 0; s >>= 1) {
            if (t < s) {
                if (bw[t + s] > bw[t] || (bw[t + s] == bw[t] && bidx[t + s] < bidx[t])) {
                    bw[t] = bw[t + s]; bidx[t] = bidx[t + s];
                }
            }
            __syncthreads();
        }
        if (t == 0) {
            int f = bidx[0];
            int l = (f >> 2) & 3;
            out[((size_t)row * 30 + j) * 2 + 0] =
                loc[(size_t)row * 256 + f * 2 + 0] / svr[(b * L_LV + l) * 2 + 0];
            out[((size_t)row * 30 + j) * 2 + 1] =
                loc[(size_t)row * 256 + f * 2 + 1] / svr[(b * L_LV + l) * 2 + 1];
            w[f] = -3.4e38f;
        }
        __syncthreads();
    }
}

// ---------------- host-side orchestration ----------------
static inline void gemm(hipStream_t s, const float* A, const float* W, const float* b,
                        void* C, int M, int Nout, int K, int ldc, int relu, int half_out)
{
    dim3 g((Nout + BN - 1) / BN, (M + BM - 1) / BM);
    gemm_wmma_kernel<<<g, 256, 0, s>>>(A, W, b, (float*)C, M, Nout, K, ldc, relu, half_out);
}

extern "C" void kernel_launch(void* const* d_in, const int* in_sizes, int n_in,
                              void* d_out, int out_size, void* d_ws, size_t ws_size,
                              hipStream_t stream)
{
    (void)in_sizes; (void)n_in; (void)out_size; (void)ws_size;
    const float* tgt  = (const float*)d_in[0];
    const float* refp = (const float*)d_in[1];
    const float* src  = (const float*)d_in[2];
    const float* qpos = (const float*)d_in[3];
    const float* svr  = (const float*)d_in[4];
    // params leaves in dict order: indices 5..26
    const float* PRM[22];
    for (int i = 0; i < 22; ++i) PRM[i] = (const float*)d_in[5 + i];
    // 0 sa_in_w 1 sa_in_b 2 sa_out_w 3 sa_out_b 4 off_w 5 off_b 6 aw_w 7 aw_b
    // 8 val_w 9 val_b 10 out_w 11 out_b 12 n1_g 13 n1_b 14 n2_g 15 n2_b
    // 16 n3_g 17 n3_b 18 l1_w 19 l1_b 20 l2_w 21 l2_b

    // workspace layout (floats)
    float* base = (float*)d_ws;
    size_t o = 0;
    float* outbuf = base + o; o += (size_t)M_Q * DM;        // running decoder state
    float* qbuf   = base + o; o += (size_t)M_Q * DM;        // output + query_pos
    float* qkv    = base + o; o += (size_t)M_Q * 768;       // self-attn projections
    float* satmp  = base + o; o += (size_t)M_Q * DM;        // attn output (pre-proj)
    float* delta  = base + o; o += (size_t)M_Q * DM;        // sa_out / ca_out / ff2
    float* offb   = base + o; o += (size_t)M_Q * 256;       // sampling offsets
    float* awb    = base + o; o += (size_t)M_Q * 128;       // attn weights
    float* locb   = base + o; o += (size_t)M_Q * 256;       // sampling locations
    float* cain   = base + o; o += (size_t)M_Q * DM;        // deform-attn pre-proj
    float* ff1    = base + o; o += (size_t)M_Q * FFH;       // FFN hidden
    _Float16* value = (_Float16*)(base + o);                // (N,S_IN,256) f16

    const int nQ = M_Q * DM;
    // output = tgt
    copy_kernel<<<(nQ + 255) / 256, 256, 0, stream>>>(tgt, outbuf, nQ);

    for (int l = 0; l < LYR; ++l) {
        const float* sa_in_w  = PRM[0]  + (size_t)l * 768 * DM;
        const float* sa_in_b  = PRM[1]  + (size_t)l * 768;
        const float* sa_out_w = PRM[2]  + (size_t)l * DM * DM;
        const float* sa_out_b = PRM[3]  + (size_t)l * DM;
        const float* off_w    = PRM[4]  + (size_t)l * 256 * DM;
        const float* off_b    = PRM[5]  + (size_t)l * 256;
        const float* aw_w     = PRM[6]  + (size_t)l * 128 * DM;
        const float* aw_b     = PRM[7]  + (size_t)l * 128;
        const float* val_w    = PRM[8]  + (size_t)l * DM * DM;
        const float* val_b    = PRM[9]  + (size_t)l * DM;
        const float* out_w    = PRM[10] + (size_t)l * DM * DM;
        const float* out_b    = PRM[11] + (size_t)l * DM;
        const float* n1_g     = PRM[12] + (size_t)l * DM;
        const float* n1_b     = PRM[13] + (size_t)l * DM;
        const float* n2_g     = PRM[14] + (size_t)l * DM;
        const float* n2_b     = PRM[15] + (size_t)l * DM;
        const float* n3_g     = PRM[16] + (size_t)l * DM;
        const float* n3_b     = PRM[17] + (size_t)l * DM;
        const float* l1_w     = PRM[18] + (size_t)l * FFH * DM;
        const float* l1_b     = PRM[19] + (size_t)l * FFH;
        const float* l2_w     = PRM[20] + (size_t)l * DM * FFH;
        const float* l2_b     = PRM[21] + (size_t)l * DM;

        // ---- self attention ----
        add_kernel<<<(nQ + 255) / 256, 256, 0, stream>>>(outbuf, qpos, qbuf, nQ);
        // q,k projections from (output+query_pos); v projection from output
        gemm(stream, qbuf,   sa_in_w,            sa_in_b,       qkv,       M_Q, 512, DM, 768, 0, 0);
        gemm(stream, outbuf, sa_in_w + 512 * DM, sa_in_b + 512, qkv + 512, M_Q, 256, DM, 768, 0, 0);
        self_attn_kernel<<<dim3(LQ, NH, N_B), 128, 0, stream>>>(qkv, satmp);
        gemm(stream, satmp, sa_out_w, sa_out_b, delta, M_Q, DM, DM, DM, 0, 0);
        add_ln_kernel<<<M_Q, 256, 0, stream>>>(outbuf, delta, n2_g, n2_b, outbuf);

        // ---- deformable cross attention ----
        add_kernel<<<(nQ + 255) / 256, 256, 0, stream>>>(outbuf, qpos, qbuf, nQ);
        gemm(stream, src,  val_w, val_b, value, M_V, DM, DM, DM, 0, 1);    // f16 output
        gemm(stream, qbuf, off_w, off_b, offb,  M_Q, 256, DM, 256, 0, 0);
        gemm(stream, qbuf, aw_w,  aw_b,  awb,   M_Q, 128, DM, 128, 0, 0);
        aw_softmax_kernel<<<M_Q, 128, 0, stream>>>(awb);
        loc_kernel<<<M_Q, 256, 0, stream>>>(offb, refp, svr, locb);
        deform_sample_kernel<<<M_Q, 256, 0, stream>>>(value, locb, awb, cain);
        gemm(stream, cain, out_w, out_b, delta, M_Q, DM, DM, DM, 0, 0);
        add_ln_kernel<<<M_Q, 256, 0, stream>>>(outbuf, delta, n1_g, n1_b, outbuf);

        // ---- FFN ----
        gemm(stream, outbuf, l1_w, l1_b, ff1,   M_Q, FFH, DM, FFH, 1, 0);
        gemm(stream, ff1,    l2_w, l2_b, delta, M_Q, DM, FFH, DM, 0, 0);
        add_ln_kernel<<<M_Q, 256, 0, stream>>>(outbuf, delta, n3_g, n3_b, outbuf);

        // ---- top-k sampling-location gather (only last layer's survives) ----
        if (l == LYR - 1) {
            float* samples_out = (float*)d_out + (size_t)M_Q * DM + (size_t)N_B * LQ * 2;
            topk_kernel<<<M_Q, 128, 0, stream>>>(awb, locb, svr, samples_out);
        }
    }

    // outputs: [output (2400*256) | reference_points (4800) | samples_keep (144000)]
    copy_kernel<<<(nQ + 255) / 256, 256, 0, stream>>>(outbuf, (float*)d_out, nQ);
    copy_kernel<<<(N_B * LQ * 2 + 255) / 256, 256, 0, stream>>>(
        refp, (float*)d_out + (size_t)M_Q * DM, N_B * LQ * 2);
}
